// Encoder_23974507446482
// MI455X (gfx1250) — matrix operand order repaired and checked by reading, not compiled, not run on hardware
//
#include <hip/hip_runtime.h>
#include <hip/hip_bf16.h>
#include <math.h>

// ---------------------------------------------------------------------------
// Types for CDNA5 (gfx1250, wave32)
// ---------------------------------------------------------------------------
typedef __bf16 bf16_t;
typedef __attribute__((ext_vector_type(16))) __bf16        v16bf;
typedef __attribute__((ext_vector_type(8)))  float         v8f;
typedef __attribute__((ext_vector_type(4)))  unsigned int  u32x4;
typedef __attribute__((ext_vector_type(8)))  int           i32x8;
typedef __attribute__((ext_vector_type(4)))  int           i32x4;

#define MODEL_D 1024
#define SEQ_S   2048
#define HEADS   16
#define DK      64
#define ROWS    4096          // B*S
#define FFN_F   4096

__device__ __forceinline__ v8f wmma_bf16(v16bf a, v16bf b, v8f c) {
  return __builtin_amdgcn_wmma_f32_16x16x32_bf16(false, a, false, b, (short)0, c,
                                                 false, false);
}

// ---------------------------------------------------------------------------
// TDM: DMA one 128-row x 32-elem bf16 tile (row stride = stride_elems) from
// global into LDS at lds_byte_off, padding 16B after each 64B row so the LDS
// row stride is 80B (= 40 bf16, 16B-aligned for ds_load_b128 fragments).
// D# packing per CDNA5 ISA.
// clang-23 toolchain: 6-arg builtin (g0 u32x4, g1 i32x8, i32x4, i32x4, i32x8, cpol)
// ---------------------------------------------------------------------------
#define LDT 40   // LDS row stride in bf16 elements

__device__ __forceinline__ void tdm_load_tile(unsigned lds_byte_off,
                                              const bf16_t* gsrc,
                                              int stride_elems)
{
  const unsigned long long ga = (unsigned long long)(size_t)gsrc;
  u32x4 g0;
  g0[0] = 1u;                                           // count=1, user mode
  g0[1] = lds_byte_off;                                 // lds_addr
  g0[2] = (unsigned)ga;                                 // global_addr[31:0]
  g0[3] = (unsigned)((ga >> 32) & 0x01FFFFFFu) | (2u << 30);  // addr[56:32]|type=2
  i32x8 g1;
  g1[0] = (1 << 16)       // data_size = 2 bytes
        | (1 << 20)       // pad_enable
        | (3 << 22)       // pad_interval: 16 DWORDs (64B)
        | (3 << 25);      // pad_amount: 4 DWORDs (16B)
  g1[1] = (int)(32u << 16);        // tensor_dim0[15:0] = 32 @ bits 63:48
  g1[2] = (int)(128u << 16);       // tensor_dim1[15:0] = 128 @ bits 95:80
  g1[3] = (int)(32u << 16);        // tile_dim0 = 32 @ bits 127:112
  g1[4] = 128;                     // tile_dim1 = 128 rows, tile_dim2 = 0
  g1[5] = stride_elems;            // tensor_dim0_stride[31:0]
  g1[6] = 0;
  g1[7] = 0;
  i32x4 gz4 = {0, 0, 0, 0};
  i32x8 gz8 = {0, 0, 0, 0, 0, 0, 0, 0};
  __builtin_amdgcn_tensor_load_to_lds(g0, g1, gz4, gz4, gz8, 0);
}

// ---------------------------------------------------------------------------
// GEMM: out[M,N] = act(A[M,K](bf16) @ Wt[N,K](bf16)^T + bias) (+ resid)
// FLAGS: bit0 = ReLU, bit1 = add fp32 residual.  OUTBF: bf16 vs fp32 output.
// 256 thr = 8 waves; 128x128 tile; K-step 32; TDM double-buffered LDS staging.
// ---------------------------------------------------------------------------
template<int FLAGS, bool OUTBF>
__global__ __launch_bounds__(256)
void gemm_tdm_kernel(const bf16_t* __restrict__ A, const bf16_t* __restrict__ Wt,
                     const float* __restrict__ bias,
                     const float* __restrict__ resid,
                     float* __restrict__ outF, bf16_t* __restrict__ outB,
                     int M, int N, int K)
{
  __shared__ bf16_t lA[2][128 * LDT];
  __shared__ bf16_t lB[2][128 * LDT];

  const int tid  = threadIdx.x;
  const int lane = tid & 31;
  const int wave = tid >> 5;               // 0..7
  const int half = lane >> 4;
  const int l16  = lane & 15;
  const int wm   = (wave >> 2) * 64;
  const int wn   = (wave & 3) * 32;
  const int bm   = blockIdx.x * 128;
  const int bn   = blockIdx.y * 128;
  const int nk   = K >> 5;

  v8f acc[4][2] = {};

  if (wave == 0) {     // one wave drives the TDM; EXEC ignored by tensor ops
    tdm_load_tile((unsigned)(size_t)(void*)&lA[0][0], A  + (size_t)bm * K, K);
    tdm_load_tile((unsigned)(size_t)(void*)&lB[0][0], Wt + (size_t)bn * K, K);
  }

  for (int i = 0; i < nk; ++i) {
    const int cur = i & 1;
    if (wave == 0) {
      if (i + 1 < nk) {
        const int k0 = (i + 1) << 5;
        tdm_load_tile((unsigned)(size_t)(void*)&lA[cur ^ 1][0],
                      A  + (size_t)bm * K + k0, K);
        tdm_load_tile((unsigned)(size_t)(void*)&lB[cur ^ 1][0],
                      Wt + (size_t)bn * K + k0, K);
        __builtin_amdgcn_s_wait_tensorcnt(2);   // tile i done; i+1 in flight
      } else {
        __builtin_amdgcn_s_wait_tensorcnt(0);
      }
    }
    __syncthreads();                            // publish tile i to all waves

    const bf16_t* sA = &lA[cur][0];
    const bf16_t* sB = &lB[cur][0];
    v16bf af[4], bfg[2];
    const int kA = half * 8;
    #pragma unroll
    for (int mi = 0; mi < 4; ++mi) {
      const bf16_t* p = &sA[(wm + mi * 16 + l16) * LDT + kA];
      #pragma unroll
      for (int e = 0; e < 8; ++e) { af[mi][e] = p[e]; af[mi][8 + e] = p[16 + e]; }
    }
    const int kB = half * 8;
    #pragma unroll
    for (int ni = 0; ni < 2; ++ni) {
      const bf16_t* p = &sB[(wn + ni * 16 + l16) * LDT + kB];
      #pragma unroll
      for (int e = 0; e < 8; ++e) { bfg[ni][e] = p[e]; bfg[ni][8 + e] = p[16 + e]; }
    }
    #pragma unroll
    for (int mi = 0; mi < 4; ++mi)
      #pragma unroll
      for (int ni = 0; ni < 2; ++ni)
        acc[mi][ni] = wmma_bf16(af[mi], bfg[ni], acc[mi][ni]);

    __syncthreads();                            // done reading before overwrite
  }

  // ---- branch-free epilogue ----
  #pragma unroll
  for (int mi = 0; mi < 4; ++mi) {
    #pragma unroll
    for (int ni = 0; ni < 2; ++ni) {
      const int col = bn + wn + ni * 16 + l16;
      const float bv = bias[col];
      #pragma unroll
      for (int v = 0; v < 8; ++v) {
        const int row = bm + wm + mi * 16 + v + half * 8;
        float x = acc[mi][ni][v] + bv;
        if constexpr ((FLAGS & 2) != 0) x += resid[(size_t)row * N + col];
        if constexpr ((FLAGS & 1) != 0) x = fmaxf(x, 0.0f);
        if constexpr (OUTBF) outB[(size_t)row * N + col] = (bf16_t)x;
        else                 outF[(size_t)row * N + col] = x;
      }
    }
  }
}

// ---------------------------------------------------------------------------
// Flash attention on bf16 Q/K/V: per (b,h) and 64-query tile; 4 waves,
// each wave owns 16 queries; online softmax; all matmuls via WMMA bf16.
// ---------------------------------------------------------------------------
#define VLD 72   // padded LDS stride (144B, multiple of 16)

__global__ __launch_bounds__(128)
void attn_kernel(const bf16_t* __restrict__ Q, const bf16_t* __restrict__ K,
                 const bf16_t* __restrict__ V, const int* __restrict__ mask,
                 bf16_t* __restrict__ ctx)
{
  const int bh   = blockIdx.y;
  const int b    = bh >> 4;
  const int h    = bh & 15;
  const int hoff = h * DK;
  const int wave = threadIdx.x >> 5;
  const int lane = threadIdx.x & 31;
  const int half = lane >> 4;
  const int l16  = lane & 15;
  const int q0   = blockIdx.x * 64 + wave * 16;
  const size_t rowBase = (size_t)b * SEQ_S;

  __shared__ bf16_t Vt[DK * VLD];        // [dk][key] transposed V tile
  __shared__ bf16_t Pl[4][16 * VLD];     // per-wave P tile [q][key]

  v16bf qf[2];
  {
    const bf16_t* qp = Q + (rowBase + q0 + l16) * MODEL_D + hoff;
    #pragma unroll
    for (int s = 0; s < 2; ++s) {
      const int kb = s * 32 + half * 8;
      #pragma unroll
      for (int e = 0; e < 8; ++e) {
        qf[s][e]     = qp[kb + e];
        qf[s][8 + e] = qp[kb + 16 + e];
      }
    }
  }

  v8f o[4] = {};
  float rmax[8], rsum[8];
  #pragma unroll
  for (int v = 0; v < 8; ++v) { rmax[v] = -1e30f; rsum[v] = 0.0f; }

  const int* mrow = mask + b * SEQ_S;

  for (int kb0 = 0; kb0 < SEQ_S; kb0 += 64) {
    __syncthreads();
    {   // cooperative transposed V tile load (bf16 -> bf16)
      const int c  = (threadIdx.x & 15) * 4;
      const int r0 = threadIdx.x >> 4;
      #pragma unroll
      for (int rr = 0; rr < 8; ++rr) {
        const int r = r0 + rr * 8;
        const bf16_t* src = V + (rowBase + kb0 + r) * MODEL_D + hoff + c;
        #pragma unroll
        for (int j = 0; j < 4; ++j) Vt[(c + j) * VLD + r] = src[j];
      }
    }
    __syncthreads();

    // S = Q @ K^T : B fragments are contiguous bf16 reads from global K
    v8f sc[4] = {};
    #pragma unroll
    for (int ks = 0; ks < 2; ++ks) {
      #pragma unroll
      for (int nt = 0; nt < 4; ++nt) {
        const int key = kb0 + nt * 16 + l16;
        const bf16_t* kp = K + (rowBase + key) * MODEL_D + hoff + ks * 32 + half * 8;
        v16bf bfrag;
        #pragma unroll
        for (int e = 0; e < 8; ++e) { bfrag[e] = kp[e]; bfrag[8 + e] = kp[16 + e]; }
        sc[nt] = wmma_bf16(qf[ks], bfrag, sc[nt]);
      }
    }

    float madd[4];
    #pragma unroll
    for (int nt = 0; nt < 4; ++nt)
      madd[nt] = (mrow[kb0 + nt * 16 + l16] == 0) ? -1e9f : 0.0f;
    #pragma unroll
    for (int nt = 0; nt < 4; ++nt)
      #pragma unroll
      for (int v = 0; v < 8; ++v)
        sc[nt][v] = sc[nt][v] * 0.125f + madd[nt];

    // online softmax (row reductions stay within 16-lane halves)
    float corr[8];
    #pragma unroll
    for (int v = 0; v < 8; ++v) {
      float x = sc[0][v];
      #pragma unroll
      for (int nt = 1; nt < 4; ++nt) x = fmaxf(x, sc[nt][v]);
      #pragma unroll
      for (int m = 1; m < 16; m <<= 1) x = fmaxf(x, __shfl_xor(x, m, 32));
      const float mn = fmaxf(rmax[v], x);
      corr[v] = __expf(rmax[v] - mn);
      rmax[v] = mn;
      float accp = 0.0f;
      #pragma unroll
      for (int nt = 0; nt < 4; ++nt) {
        const float p = __expf(sc[nt][v] - mn);
        sc[nt][v] = p;
        accp += p;
      }
      #pragma unroll
      for (int m = 1; m < 16; m <<= 1) accp += __shfl_xor(accp, m, 32);
      rsum[v] = rsum[v] * corr[v] + accp;
    }

    // P -> per-wave LDS (C layout -> A layout re-staging)
    bf16_t* pw = &Pl[wave][0];
    #pragma unroll
    for (int nt = 0; nt < 4; ++nt)
      #pragma unroll
      for (int v = 0; v < 8; ++v)
        pw[(v + 8 * half) * VLD + nt * 16 + l16] = (bf16_t)sc[nt][v];

    v16bf pf[2];
    #pragma unroll
    for (int ks = 0; ks < 2; ++ks) {
      const bf16_t* pp = pw + l16 * VLD + ks * 32 + half * 8;
      #pragma unroll
      for (int e = 0; e < 8; ++e) { pf[ks][e] = pp[e]; pf[ks][8 + e] = pp[16 + e]; }
    }

    // O = O*corr + P @ V
    #pragma unroll
    for (int dt = 0; dt < 4; ++dt) {
      #pragma unroll
      for (int v = 0; v < 8; ++v) o[dt][v] *= corr[v];
      #pragma unroll
      for (int ks = 0; ks < 2; ++ks) {
        const bf16_t* vp = &Vt[(dt * 16 + l16) * VLD + ks * 32 + half * 8];
        v16bf vfrag;
        #pragma unroll
        for (int e = 0; e < 8; ++e) { vfrag[e] = vp[e]; vfrag[8 + e] = vp[16 + e]; }
        o[dt] = wmma_bf16(pf[ks], vfrag, o[dt]);
      }
    }
  }

  #pragma unroll
  for (int v = 0; v < 8; ++v) {
    const float inv = 1.0f / rsum[v];
    const int row = q0 + v + 8 * half;
    #pragma unroll
    for (int dt = 0; dt < 4; ++dt)
      ctx[(rowBase + row) * MODEL_D + hoff + dt * 16 + l16] =
          (bf16_t)(o[dt][v] * inv);
  }
}

// ---------------------------------------------------------------------------
// Row LayerNorm over D=1024; optional extra bf16 output for next GEMM.
// ---------------------------------------------------------------------------
__global__ __launch_bounds__(256)
void layernorm_kernel(const float* __restrict__ in, const float* __restrict__ g,
                      const float* __restrict__ beta, float* __restrict__ out,
                      bf16_t* __restrict__ outb)
{
  const int row = blockIdx.x;
  const float* x = in + (size_t)row * MODEL_D;
  const int t = threadIdx.x;
  const int lane = t & 31, wave = t >> 5;

  float v[4], s = 0.0f, s2 = 0.0f;
  #pragma unroll
  for (int i = 0; i < 4; ++i) {
    const float a = x[t + i * 256];
    v[i] = a; s += a; s2 += a * a;
  }
  #pragma unroll
  for (int m = 1; m < 32; m <<= 1) {
    s  += __shfl_xor(s,  m, 32);
    s2 += __shfl_xor(s2, m, 32);
  }
  __shared__ float rs[8], rs2[8];
  __shared__ float mu_s, rstd_s;
  if (lane == 0) { rs[wave] = s; rs2[wave] = s2; }
  __syncthreads();
  if (t == 0) {
    float ts = 0.0f, ts2 = 0.0f;
    #pragma unroll
    for (int i = 0; i < 8; ++i) { ts += rs[i]; ts2 += rs2[i]; }
    const float mu = ts * (1.0f / MODEL_D);
    const float var = ts2 * (1.0f / MODEL_D) - mu * mu;
    mu_s = mu;
    rstd_s = rsqrtf(var + 1e-5f);
  }
  __syncthreads();
  const float mu = mu_s, rstd = rstd_s;
  float* y = out + (size_t)row * MODEL_D;
  #pragma unroll
  for (int i = 0; i < 4; ++i) {
    const int c = t + i * 256;
    const float r = (v[i] - mu) * rstd * g[c] + beta[c];
    y[c] = r;
    if (outb) outb[(size_t)row * MODEL_D + c] = (bf16_t)r;
  }
}

// ---------------------------------------------------------------------------
// One-time fp32 -> bf16 conversion; and fp32[K,N] -> bf16[N,K] transpose.
// ---------------------------------------------------------------------------
__global__ __launch_bounds__(256)
void cvt_bf16_kernel(const float* __restrict__ in, bf16_t* __restrict__ out)
{
  const int i = blockIdx.x * 1024 + threadIdx.x * 4;
  #pragma unroll
  for (int j = 0; j < 4; ++j) out[i + j] = (bf16_t)in[i + j];
}

__global__ __launch_bounds__(256)
void transpose_cvt_kernel(const float* __restrict__ W, bf16_t* __restrict__ Wt,
                          int K, int N)
{
  __shared__ bf16_t t[32][33];
  const int tx = threadIdx.x & 31, ty = threadIdx.x >> 5;
  const int n0 = blockIdx.x * 32, k0 = blockIdx.y * 32;
  #pragma unroll
  for (int j = 0; j < 4; ++j)
    t[ty + 8 * j][tx] = (bf16_t)W[(size_t)(k0 + ty + 8 * j) * N + n0 + tx];
  __syncthreads();
  #pragma unroll
  for (int j = 0; j < 4; ++j)
    Wt[(size_t)(n0 + ty + 8 * j) * K + k0 + tx] = t[tx][ty + 8 * j];
}

// ---------------------------------------------------------------------------
// Launch pipeline
// ---------------------------------------------------------------------------
extern "C" void kernel_launch(void* const* d_in, const int* in_sizes, int n_in,
                              void* d_out, int out_size, void* d_ws, size_t ws_size,
                              hipStream_t stream)
{
  const float* inp  = (const float*)d_in[0];
  const int*   mask = (const int*)  d_in[1];
  const float* Wq   = (const float*)d_in[2];
  const float* bq   = (const float*)d_in[3];
  const float* Wk   = (const float*)d_in[4];
  const float* bk   = (const float*)d_in[5];
  const float* Wv   = (const float*)d_in[6];
  const float* bv   = (const float*)d_in[7];
  const float* Wo   = (const float*)d_in[8];
  const float* bo   = (const float*)d_in[9];
  const float* ln_g = (const float*)d_in[10];
  const float* ln_b = (const float*)d_in[11];
  const float* W1   = (const float*)d_in[12];
  const float* b1   = (const float*)d_in[13];
  const float* W2   = (const float*)d_in[14];
  const float* b2   = (const float*)d_in[15];
  float* outp = (float*)d_out;

  char* ws = (char*)d_ws;
  const size_t MB = 1024 * 1024;
  bf16_t* inp_bf = (bf16_t*)(ws +  0 * MB);   //  8 MB
  bf16_t* WqT    = (bf16_t*)(ws +  8 * MB);   //  2 MB
  bf16_t* WkT    = (bf16_t*)(ws + 10 * MB);   //  2 MB
  bf16_t* WvT    = (bf16_t*)(ws + 12 * MB);   //  2 MB
  bf16_t* WoT    = (bf16_t*)(ws + 14 * MB);   //  2 MB
  bf16_t* W1T    = (bf16_t*)(ws + 16 * MB);   //  8 MB
  bf16_t* W2T    = (bf16_t*)(ws + 24 * MB);   //  8 MB
  bf16_t* Qb     = (bf16_t*)(ws + 32 * MB);   //  8 MB
  bf16_t* Kb     = (bf16_t*)(ws + 40 * MB);   //  8 MB
  bf16_t* Vb     = (bf16_t*)(ws + 48 * MB);   //  8 MB
  bf16_t* ctxb   = (bf16_t*)(ws + 56 * MB);   //  8 MB
  float*  y1     = (float*) (ws + 64 * MB);   // 16 MB
  float*  xln    = (float*) (ws + 80 * MB);   // 16 MB
  bf16_t* xlnb   = (bf16_t*)(ws + 96 * MB);   //  8 MB
  bf16_t* hffn   = (bf16_t*)(ws + 32 * MB);   // 32 MB, reuse Qb..ctxb
  float*  y2     = (float*) (ws + 64 * MB);   // 16 MB, reuse y1

  dim3 blk(256);
  dim3 gD(ROWS / 128, MODEL_D / 128);
  dim3 gF(ROWS / 128, FFN_F / 128);

  // one-time conversions
  cvt_bf16_kernel<<<dim3(ROWS * MODEL_D / 1024), blk, 0, stream>>>(inp, inp_bf);
  transpose_cvt_kernel<<<dim3(32, 32), blk, 0, stream>>>(Wq, WqT, 1024, 1024);
  transpose_cvt_kernel<<<dim3(32, 32), blk, 0, stream>>>(Wk, WkT, 1024, 1024);
  transpose_cvt_kernel<<<dim3(32, 32), blk, 0, stream>>>(Wv, WvT, 1024, 1024);
  transpose_cvt_kernel<<<dim3(32, 32), blk, 0, stream>>>(Wo, WoT, 1024, 1024);
  transpose_cvt_kernel<<<dim3(128, 32), blk, 0, stream>>>(W1, W1T, 1024, 4096);
  transpose_cvt_kernel<<<dim3(32, 128), blk, 0, stream>>>(W2, W2T, 4096, 1024);

  // QKV projections (bf16 out)
  gemm_tdm_kernel<0, true><<<gD, blk, 0, stream>>>(inp_bf, WqT, bq, nullptr,
                                                   nullptr, Qb, ROWS, MODEL_D, MODEL_D);
  gemm_tdm_kernel<0, true><<<gD, blk, 0, stream>>>(inp_bf, WkT, bk, nullptr,
                                                   nullptr, Kb, ROWS, MODEL_D, MODEL_D);
  gemm_tdm_kernel<0, true><<<gD, blk, 0, stream>>>(inp_bf, WvT, bv, nullptr,
                                                   nullptr, Vb, ROWS, MODEL_D, MODEL_D);

  // attention
  attn_kernel<<<dim3(SEQ_S / 64, 2 * HEADS), dim3(128), 0, stream>>>(
      Qb, Kb, Vb, mask, ctxb);

  // output projection + residual(inp), then LN (fp32 + bf16 outputs)
  gemm_tdm_kernel<2, false><<<gD, blk, 0, stream>>>(ctxb, WoT, bo, inp,
                                                    y1, nullptr, ROWS, MODEL_D, MODEL_D);
  layernorm_kernel<<<dim3(ROWS), blk, 0, stream>>>(y1, ln_g, ln_b, xln, xlnb);

  // FFN
  gemm_tdm_kernel<1, true><<<gF, blk, 0, stream>>>(xlnb, W1T, b1, nullptr,
                                                   nullptr, hffn, ROWS, FFN_F, MODEL_D);
  gemm_tdm_kernel<2, false><<<gD, blk, 0, stream>>>(hffn, W2T, b2, xln,
                                                    y2, nullptr, ROWS, MODEL_D, FFN_F);
  layernorm_kernel<<<dim3(ROWS), blk, 0, stream>>>(y2, ln_g, ln_b, outp, nullptr);
}
